// GraphConvolution_75892072120902
// MI455X (gfx1250) — matrix operand order repaired and checked by reading, not compiled, not run on hardware
//
#include <hip/hip_runtime.h>
#include <hip/hip_bf16.h>

#define N_NODES 55
#define N_FEAT 48
#define TILES_PER_B 4   // ceil(55/16)
#define THREADS 256
#define WAVES_PER_BLOCK (THREADS / 32)

typedef __attribute__((ext_vector_type(16))) _Float16 v16h;
typedef __attribute__((ext_vector_type(8)))  float    v8f;
typedef __attribute__((ext_vector_type(4)))  float    v4f;

// y[b,i,o] = relu( sum_f AX[b,i,f] * W[o,f] + bias[o] )
// AX[b,i,f] = (x[b,i-1,f] + x[b,i,f] + x[b,i+1,f]) / deg(i), deg = 3 - (i==0) - (i==54)
__global__ __launch_bounds__(THREADS)
void gconv_wmma_f16split(const float* __restrict__ x,
                         const float* __restrict__ W,
                         const float* __restrict__ bias,
                         float* __restrict__ y,
                         int batch) {
    __shared__ float sW[N_FEAT * N_FEAT];   // 9216 B, loaded once per block

    const int tid = threadIdx.x;
    for (int i = tid; i < N_FEAT * N_FEAT; i += THREADS) sW[i] = W[i];
    __syncthreads();

    const int lane   = tid & 31;
    const int waveId = tid >> 5;
    const int n      = lane & 15;   // N-col (B/C/D) and M-row (A) index within tile
    const int hiHalf = lane >> 4;   // which half-wave this lane is in

    // ---- Build B fragments once (kept in VGPRs for the whole kernel) ----
    // B is 32x16 (KxN): lanes 0-15 hold K in {0-7, 16-23}, lanes 16-31 hold K in {8-15, 24-31},
    // column n = lane%16. B[k][n] = W[o = nt*16+n, f = k]  (einsum 'bif,of->bio').
    v16h bhi[2][3], blo[2][3];
#pragma unroll
    for (int kt = 0; kt < 2; ++kt) {
#pragma unroll
        for (int nt = 0; nt < 3; ++nt) {
            v16h fh{}, fl{};
            const int o = nt * 16 + n;
#pragma unroll
            for (int e = 0; e < 16; ++e) {
                const int k = kt * 32 + (e < 8 ? hiHalf * 8 + e : 16 + hiHalf * 8 + (e - 8));
                float w = (k < N_FEAT) ? sW[o * N_FEAT + k] : 0.0f;
                _Float16 h = (_Float16)w;
                fh[e] = h;
                fl[e] = (_Float16)(w - (float)h);
            }
            bhi[kt][nt] = fh;
            blo[kt][nt] = fl;
        }
    }

    // per-lane bias values (column o = nt*16 + n)
    float bval[3];
#pragma unroll
    for (int nt = 0; nt < 3; ++nt) bval[nt] = bias[nt * 16 + n];

    const int       globalWave = blockIdx.x * WAVES_PER_BLOCK + waveId;
    const int       numWaves   = gridDim.x * WAVES_PER_BLOCK;
    const long long totalTiles = (long long)batch * TILES_PER_B;

    for (long long tIdx = globalWave; tIdx < totalTiles; tIdx += numWaves) {
        const int b = (int)(tIdx >> 2);
        const int t = (int)(tIdx & 3);

        // A-matrix row this lane supplies (both lane halves supply row M = lane%16).
        const int  r        = t * 16 + n;
        const bool rowValid = (r < N_NODES);
        const float coef = rowValid
            ? 1.0f / (float)(3 - (r == 0 ? 1 : 0) - (r == N_NODES - 1 ? 1 : 0))
            : 0.0f;
        const float* xb = x + (long long)b * (N_NODES * N_FEAT);
        const float* xr = xb + r * N_FEAT;

        v8f acc[3] = {};

#pragma unroll
        for (int kt = 0; kt < 2; ++kt) {
            // A layout (16-bit 16x32): lanes 0-15 hold K in {0-7, 16-23},
            // lanes 16-31 hold K in {8-15, 24-31}; elements e=0..7 -> VGPR0-3,
            // e=8..15 -> VGPR4-7. Groups of 8 consecutive K per element-half.
            v16h ah{}, al{};
#pragma unroll
            for (int g = 0; g < 2; ++g) {
                const int kstart = kt * 32 + g * 16 + hiHalf * 8;
                if (kstart < N_FEAT && rowValid) {
                    v4f s0 = *(const v4f*)(xr + kstart);
                    v4f s1 = *(const v4f*)(xr + kstart + 4);
                    if (r > 0) {
                        s0 += *(const v4f*)(xr - N_FEAT + kstart);
                        s1 += *(const v4f*)(xr - N_FEAT + kstart + 4);
                    }
                    if (r < N_NODES - 1) {
                        s0 += *(const v4f*)(xr + N_FEAT + kstart);
                        s1 += *(const v4f*)(xr + N_FEAT + kstart + 4);
                    }
                    s0 *= coef;
                    s1 *= coef;
#pragma unroll
                    for (int q = 0; q < 4; ++q) {
                        _Float16 h0 = (_Float16)s0[q];
                        _Float16 h1 = (_Float16)s1[q];
                        ah[g * 8 + q]     = h0;
                        ah[g * 8 + 4 + q] = h1;
                        al[g * 8 + q]     = (_Float16)(s0[q] - (float)h0);
                        al[g * 8 + 4 + q] = (_Float16)(s1[q] - (float)h1);
                    }
                }
            }

            // Error-compensated f16 matmul: hi*hi + lo*hi + hi*lo, f32 accumulate.
#pragma unroll
            for (int nt = 0; nt < 3; ++nt) {
                acc[nt] = __builtin_amdgcn_wmma_f32_16x16x32_f16(
                    false, ah, false, bhi[kt][nt], (short)0, acc[nt], false, false);
                acc[nt] = __builtin_amdgcn_wmma_f32_16x16x32_f16(
                    false, al, false, bhi[kt][nt], (short)0, acc[nt], false, false);
                acc[nt] = __builtin_amdgcn_wmma_f32_16x16x32_f16(
                    false, ah, false, blo[kt][nt], (short)0, acc[nt], false, false);
            }
        }

        // Epilogue. C/D layout: VGPR v holds (M = v + 8*hiHalf, N = n).
        // y is write-once (173 MB): non-temporal stores keep L2 capacity for
        // x's 3x neighbor-row reuse (x nearly fits in the 192 MB L2).
        float* yb = y + (long long)b * (N_NODES * N_FEAT);
#pragma unroll
        for (int nt = 0; nt < 3; ++nt) {
            const int o = nt * 16 + n;
#pragma unroll
            for (int v = 0; v < 8; ++v) {
                const int m = t * 16 + v + 8 * hiHalf;
                if (m < N_NODES) {
                    const float val = fmaxf(acc[nt][v] + bval[nt], 0.0f);
                    __builtin_nontemporal_store(val, &yb[m * N_FEAT + o]);
                }
            }
        }
    }
}

extern "C" void kernel_launch(void* const* d_in, const int* in_sizes, int n_in,
                              void* d_out, int out_size, void* d_ws, size_t ws_size,
                              hipStream_t stream) {
    const float* x    = (const float*)d_in[0];
    const float* W    = (const float*)d_in[1];
    const float* bias = (const float*)d_in[2];
    float*       y    = (float*)d_out;

    const int batch = in_sizes[0] / (N_NODES * N_FEAT);
    const long long tiles = (long long)batch * TILES_PER_B;

    long long blocks = (tiles + WAVES_PER_BLOCK - 1) / WAVES_PER_BLOCK;
    if (blocks > 2048) blocks = 2048;   // grid-stride; amortize W/B-fragment setup
    if (blocks < 1) blocks = 1;

    hipLaunchKernelGGL(gconv_wmma_f16split, dim3((unsigned)blocks), dim3(THREADS),
                       0, stream, x, W, bias, y, batch);
}
